// minLSTMCell_77086073028780
// MI455X (gfx1250) — matrix operand order repaired and checked by reading, not compiled, not run on hardware
//
#include <hip/hip_runtime.h>
#include <hip/hip_bf16.h>
#include <math.h>

// minLSTM on MI455X (gfx1250):
//   Pass 0: convert x_t and W from f32 -> bf16 once (streaming, ~190 MiB).
//   Pass 1: gates GEMM (M=32768, N=3072, K=1024) with v_wmma_f32_16x16x32_bf16,
//           bf16 fragments loaded directly via global_load_b128 (no cvt in loop),
//           fused gate-math epilogue storing log_f / log_values (256 MiB ws).
//   Pass 2: per-(b,h) sequential scan: cumsum + streaming cumlogsumexp.
//
// Workspace layout (d_ws): lf (128 MiB) | lv (128 MiB) | Xbf16 (64 MiB) | Wbf16 (6 MiB)

#define EPSF 1e-8f
#define BB 8
#define SS 4096
#define DD 1024
#define HH 1024
#define MM (BB * SS)   // 32768

typedef __attribute__((ext_vector_type(16))) __bf16 v16bf;
typedef __attribute__((ext_vector_type(8)))  __bf16 v8bf;
typedef __attribute__((ext_vector_type(8)))  float  v8f;

__device__ __forceinline__ float softplusf(float x) {
    return fmaxf(x, 0.0f) + log1pf(__expf(-fabsf(x)));
}

// ---------------------------------------------------------------------------
// Pass 0: f32 -> bf16 elementwise convert, 8 elements per thread.
// ---------------------------------------------------------------------------
__global__ __launch_bounds__(256)
void cvt_f32_to_bf16(const float* __restrict__ src, __bf16* __restrict__ dst)
{
    const size_t i = (size_t)blockIdx.x * blockDim.x + threadIdx.x;
    const float4* s = (const float4*)src;
    float4 a = s[2 * i], b = s[2 * i + 1];
    v8bf o;
    o[0] = (__bf16)a.x; o[1] = (__bf16)a.y; o[2] = (__bf16)a.z; o[3] = (__bf16)a.w;
    o[4] = (__bf16)b.x; o[5] = (__bf16)b.y; o[6] = (__bf16)b.z; o[7] = (__bf16)b.w;
    *(v8bf*)(dst + 8 * i) = o;
}

// ---------------------------------------------------------------------------
// Pass 1: gates = x @ W^T.  block = 128 threads (4 waves).
//   Block tile: 128 M-rows x 16 h-channels; each wave owns 2 M-tiles (32 rows)
//   against the 3 gate column tiles (f: n=h, i: n=H+h, hidden: n=2H+h)
//   -> 6 WMMAs per K-step of 32, B fragments reused by both M-tiles.
//   Fragment layouts used here (16-bit; contiguous k, not the ISA 7.12.2 table order):
//     A lane l: row (l&15); K = 16 contiguous   (+16 for lanes 16..31)
//     B lane l: col (l&15); K = 16 contiguous   (+16 for lanes 16..31)
//   Both K-contiguous in bf16 memory -> pairs of 16 B global_load_b128.
// ---------------------------------------------------------------------------
__global__ __launch_bounds__(128)
void gates_gemm_wmma(const __bf16* __restrict__ Xb, const __bf16* __restrict__ Wb,
                     float* __restrict__ lf_out, float* __restrict__ lv_out)
{
    const int lane = threadIdx.x & 31;
    const int wave = threadIdx.x >> 5;
    const int half = lane >> 4;
    const int l15  = lane & 15;

    const int m_base = blockIdx.x * 128 + wave * 32;   // wave owns rows [m_base, m_base+32)
    const int h_base = blockIdx.y * 16;

    const __bf16* arow0 = Xb + (size_t)(m_base + l15) * DD + half * 16;
    const __bf16* arow1 = arow0 + (size_t)16 * DD;
    const __bf16* br0   = Wb + (size_t)(0 * HH + h_base + l15) * DD + half * 16;
    const __bf16* br1   = br0 + (size_t)HH * DD;
    const __bf16* br2   = br1 + (size_t)HH * DD;

    v8f accF0 = {}, accI0 = {}, accH0 = {};
    v8f accF1 = {}, accI1 = {}, accH1 = {};

    auto loadA = [](const __bf16* p) -> v16bf {   // 16 contiguous elems
        uint4 lo = *(const uint4*)p;
        uint4 hi = *(const uint4*)(p + 8);
        v16bf r;
        __builtin_memcpy(&r, &lo, 16);
        __builtin_memcpy(reinterpret_cast<char*>(&r) + 16, &hi, 16);
        return r;
    };
    auto loadB = [](const __bf16* p) -> v16bf {   // 16 contiguous elems
        uint4 lo = *(const uint4*)p;
        uint4 hi = *(const uint4*)(p + 8);
        v16bf r;
        __builtin_memcpy(&r, &lo, 16);
        __builtin_memcpy(reinterpret_cast<char*>(&r) + 16, &hi, 16);
        return r;
    };

    for (int k = 0; k < DD; k += 32) {
        if (k + 32 < DD) {   // near-scope prefetch of next K chunk
            __builtin_prefetch(arow0 + k + 32, 0, 3);
            __builtin_prefetch(arow1 + k + 32, 0, 3);
            __builtin_prefetch(br0   + k + 32, 0, 3);
            __builtin_prefetch(br1   + k + 32, 0, 3);
            __builtin_prefetch(br2   + k + 32, 0, 3);
        }

        v16bf A0 = loadA(arow0 + k);
        v16bf A1 = loadA(arow1 + k);
        v16bf BF = loadB(br0 + k);
        v16bf BI = loadB(br1 + k);
        v16bf BH = loadB(br2 + k);

        accF0 = __builtin_amdgcn_wmma_f32_16x16x32_bf16(false, A0, false, BF, (short)0, accF0, false, false);
        accF1 = __builtin_amdgcn_wmma_f32_16x16x32_bf16(false, A1, false, BF, (short)0, accF1, false, false);
        accI0 = __builtin_amdgcn_wmma_f32_16x16x32_bf16(false, A0, false, BI, (short)0, accI0, false, false);
        accI1 = __builtin_amdgcn_wmma_f32_16x16x32_bf16(false, A1, false, BI, (short)0, accI1, false, false);
        accH0 = __builtin_amdgcn_wmma_f32_16x16x32_bf16(false, A0, false, BH, (short)0, accH0, false, false);
        accH1 = __builtin_amdgcn_wmma_f32_16x16x32_bf16(false, A1, false, BH, (short)0, accH1, false, false);
    }

    // ---- fused gate-math epilogue ----
    // C layout: VGPR r -> row M = r + half*8 (lanes 0-15 / 16-31), col = l15.
    const int h = h_base + l15;
    for (int t = 0; t < 2; ++t) {
        const v8f& aF = t ? accF1 : accF0;
        const v8f& aI = t ? accI1 : accI0;
        const v8f& aH = t ? accH1 : accH0;
        for (int r = 0; r < 8; ++r) {
            const int m = m_base + t * 16 + r + half * 8;
            const size_t idx = (size_t)m * HH + h;

            const float fg = aF[r];
            const float ig = aI[r];
            const float hd = aH[r];

            const float diff  = softplusf(-fg) - softplusf(-ig);
            const float log_f = -softplusf(diff);
            const float log_i = -softplusf(-diff);
            const float log_g = (hd >= 0.0f) ? __logf(hd + 0.5f + EPSF)
                                             : (-softplusf(-hd) + EPSF);

            lf_out[idx] = log_f;            // log_coeffs
            lv_out[idx] = log_i + log_g;    // log_values
        }
    }
}

// ---------------------------------------------------------------------------
// Pass 2: per-(b,h) sequential scan over S.
//   a_star = cumsum(log_f + eps); lse = cumlogsumexp(log_v - a_star)
//   h_t = exp(a_star + lse). Streaming (max,sum) carry; coalesced over h.
// ---------------------------------------------------------------------------
__global__ __launch_bounds__(256)
void minlstm_scan(const float* __restrict__ lf, const float* __restrict__ lv,
                  float* __restrict__ out)
{
    const int t = blockIdx.x * blockDim.x + threadIdx.x;  // 0 .. B*H-1
    const int b = t / HH;
    const int h = t % HH;
    const size_t base = (size_t)b * SS * HH + h;

    float a  = 0.0f;
    float mx = -INFINITY;
    float rs = 0.0f;

    for (int s = 0; s < SS; ++s) {
        const size_t idx = base + (size_t)s * HH;
        const float lc = lf[idx] + EPSF;
        a += lc;
        const float z = lv[idx] - a;
        if (z > mx) {
            rs = rs * __expf(mx - z) + 1.0f;
            mx = z;
        } else {
            rs += __expf(z - mx);
        }
        out[idx] = __expf(a + mx + __logf(rs));
    }
}

extern "C" void kernel_launch(void* const* d_in, const int* in_sizes, int n_in,
                              void* d_out, int out_size, void* d_ws, size_t ws_size,
                              hipStream_t stream) {
    const float* X = (const float*)d_in[0];   // (8,4096,1024) f32
    const float* W = (const float*)d_in[1];   // (3072,1024)  f32
    float* out = (float*)d_out;               // (8,4096,1024) f32

    // workspace: lf | lv | Xbf16 | Wbf16  (128 + 128 + 64 + 6 MiB)
    float*  lf = (float*)d_ws;
    float*  lv = lf + (size_t)MM * HH;
    __bf16* Xb = (__bf16*)(lv + (size_t)MM * HH);
    __bf16* Wb = Xb + (size_t)MM * DD;

    // Pass 0: precision conversion (one streaming pass each)
    cvt_f32_to_bf16<<<(MM * DD / 8) / 256, 256, 0, stream>>>(X, Xb);
    cvt_f32_to_bf16<<<(3 * HH * DD / 8) / 256, 256, 0, stream>>>(W, Wb);

    // Pass 1: WMMA GEMM + fused gate epilogue
    dim3 ggrid(MM / 128, HH / 16);            // (256, 64)
    gates_gemm_wmma<<<ggrid, 128, 0, stream>>>(Xb, Wb, lf, lv);

    // Pass 2: scan
    const int nchan = BB * HH;                // 8192
    minlstm_scan<<<nchan / 256, 256, 0, stream>>>(lf, lv, out);
}
